// MultiHeadAttention_63007170232353
// MI455X (gfx1250) — matrix operand order repaired and checked by reading, not compiled, not run on hardware
//
#include <hip/hip_runtime.h>
#include <cstdint>
#include <cstddef>

// ---------------------------------------------------------------------------
// MI455X (gfx1250) fused multi-head attention, bf16 WMMA pipeline.
//   k0 (x4): W fp32 [K,N] -> Wt bf16 [N,K]   (one-time transpose+convert)
//   k1..k3 : X@W+b GEMMs -> bf16 Q*scale [B,H,S,hd], K [B,H,S,hd], V^T [B,H,hd,S]
//            (B tile staged with global_load_async_to_lds_b128, double-buffered)
//   k4     : flash attention; row max via v_permlane16 (VALU, no LDS), row sums
//            on the matrix unit via P x ones WMMA; ctx bf16 [B,S,D]
//   k5     : ctx@Wo+bo -> fp32 out (A and B tiles both async-staged)
// Workspace: Q,K,Vt,ctx (8 MiB each) + 4 bf16 weights (2 MiB each) = 40 MiB.
// ---------------------------------------------------------------------------

#define USE_ASYNC_LDS 1

typedef __attribute__((ext_vector_type(16))) __bf16 v16bf;
typedef __attribute__((ext_vector_type(8)))  float  v8f;

union Frag { v16bf bf; uint4 q[2]; };

static __device__ __forceinline__ unsigned short f32_bf16(float f) {
  unsigned int u = __builtin_bit_cast(unsigned int, f);
  unsigned int r = 0x7FFFu + ((u >> 16) & 1u);           // round-to-nearest-even
  return (unsigned short)((u + r) >> 16);
}
static __device__ __forceinline__ v8f vzero() {
  v8f v = {0.f,0.f,0.f,0.f,0.f,0.f,0.f,0.f};
  return v;
}
static __device__ __forceinline__ v8f wmma_bf16(const Frag& a, const Frag& b, v8f c) {
  // D(16x16 f32) = A(16x32 bf16) * B(32x16 bf16) + C
  return __builtin_amdgcn_wmma_f32_16x16x32_bf16(false, a.bf, false, b.bf,
                                                 (short)0, c, false, false);
}

// ---- 16-lane xor-max reduction on the VALU via v_permlane16_b32 -------------
#if defined(__has_builtin)
#if __has_builtin(__builtin_amdgcn_permlane16)
#define HAVE_PERMLANE16 1
#endif
#endif

static __device__ __forceinline__ float max16(float m) {
#ifdef HAVE_PERMLANE16
  // selector nibbles: lane i (per 16-lane row) reads lane i^mask
  #define PL16MAX(s0, s1)                                                      \
    m = fmaxf(m, __builtin_bit_cast(float, __builtin_amdgcn_permlane16(        \
            __builtin_bit_cast(unsigned, m), __builtin_bit_cast(unsigned, m),  \
            (s0), (s1), false, false)))
  PL16MAX(0x67452301u, 0xEFCDAB89u);   // xor 1
  PL16MAX(0x54761032u, 0xDCFE98BAu);   // xor 2
  PL16MAX(0x32107654u, 0xBA98FEDCu);   // xor 4
  PL16MAX(0xFEDCBA98u, 0x76543210u);   // xor 8
  #undef PL16MAX
#else
  m = fmaxf(m, __shfl_xor(m, 1, 32));
  m = fmaxf(m, __shfl_xor(m, 2, 32));
  m = fmaxf(m, __shfl_xor(m, 4, 32));
  m = fmaxf(m, __shfl_xor(m, 8, 32));
#endif
  return m;
}

// Async copy of 16B global -> LDS, tracked by ASYNCcnt (CDNA5 path).
static __device__ __forceinline__ void async_b128(unsigned lds_byte_addr,
                                                  const void* gptr) {
#if USE_ASYNC_LDS
  asm volatile("global_load_async_to_lds_b128 %0, %1, off"
               :: "v"(lds_byte_addr),
                  "v"((unsigned long long)(uintptr_t)gptr)
               : "memory");
#endif
}
static __device__ __forceinline__ void wait_async0() {
#if USE_ASYNC_LDS
  asm volatile("s_wait_asynccnt 0x0" ::: "memory");
#endif
}
static __device__ __forceinline__ unsigned lds_addr32(const void* p) {
  // flat address of a __shared__ object: low 32 bits == LDS byte offset
  return (unsigned)(uintptr_t)p;
}

// ---------------------------------------------------------------------------
// k0: Wt[n][k] = bf16(W[k][n]); 64x64 tiles, LDS-staged so both sides coalesce.
// ---------------------------------------------------------------------------
__global__ __launch_bounds__(256) void transp_k(const float* __restrict__ W,
                                                unsigned short* __restrict__ Wt) {
  __shared__ __align__(16) unsigned short T[64][80];  // pad: 160B rows, 16B aligned
  const int tid = threadIdx.x;
  const int k0 = blockIdx.x * 64, n0 = blockIdx.y * 64;

  const int c4 = tid & 15, r = tid >> 4;              // 16 col-quads x 16 rows
  #pragma unroll
  for (int i = 0; i < 4; ++i) {
    const int kl = r + i * 16;
    const float4 v = *(const float4*)(W + (size_t)(k0 + kl) * 1024 + n0 + c4 * 4);
    T[c4 * 4 + 0][kl] = f32_bf16(v.x);
    T[c4 * 4 + 1][kl] = f32_bf16(v.y);
    T[c4 * 4 + 2][kl] = f32_bf16(v.z);
    T[c4 * 4 + 3][kl] = f32_bf16(v.w);
  }
  __syncthreads();
  const int nl = tid >> 2, sq = tid & 3;              // 64 rows x 4 segs of 32B
  #pragma unroll
  for (int i = 0; i < 2; ++i) {
    const int kk = sq * 16 + i * 8;
    *(uint4*)&Wt[(size_t)(n0 + nl) * 1024 + k0 + kk] = *(const uint4*)&T[nl][kk];
  }
}

// ---------------------------------------------------------------------------
// GEMM: Out = (A[4096,1024] @ Wt^T + bias) * outScale  (Wt is bf16 [N,K]).
// Block tile 128x128, BK=32, double-buffered LDS, async B (and async A when
// the activations are already bf16).
// MODE 0: fp32 row-major. MODE 1: bf16 [B,H,S,hd]. MODE 2: bf16 [B,H,hd,S].
// ---------------------------------------------------------------------------
template <int MODE, typename AT>
__global__ __launch_bounds__(256) void gemm_k(const AT* __restrict__ A,
                                              const unsigned short* __restrict__ Wt,
                                              const float* __restrict__ bias,
                                              void* __restrict__ OutP,
                                              float outScale) {
  __shared__ __align__(16) unsigned short As[2][128 * 32];  // [m][k] bf16
  __shared__ __align__(16) unsigned short Bs[2][128 * 32];  // [n][k] bf16

  const int tid  = threadIdx.x;
  const int n0   = blockIdx.x * 128;
  const int m0   = blockIdx.y * 128;
  const int lane = tid & 31, wv = tid >> 5;
  const int hi   = lane >> 4, l16 = lane & 15;
  const int wm   = wv >> 2,  wn = wv & 3;      // waves: 2 (M) x 4 (N)

  // copy-loader geometry: 128 rows x 64B, 2 x 16B segments per thread
  const int rowh = tid >> 1;
  const int segE = (tid & 1) * 16;             // element offset in 32-elem row

  // manual A loader (fp32 -> bf16) geometry
  const int kq = tid & 7, rg = tid >> 3;

  v8f acc[4][2];
  #pragma unroll
  for (int i = 0; i < 4; ++i)
    #pragma unroll
    for (int j = 0; j < 2; ++j) acc[i][j] = vzero();

  auto stageB = [&](int buf, int kb) {
    const unsigned short* g = Wt + (size_t)(n0 + rowh) * 1024 + kb * 32 + segE;
#if USE_ASYNC_LDS
    const unsigned l = lds_addr32(&Bs[buf][rowh * 32 + segE]);
    async_b128(l,      g);
    async_b128(l + 16, g + 8);
#else
    *(uint4*)&Bs[buf][rowh * 32 + segE]     = *(const uint4*)g;
    *(uint4*)&Bs[buf][rowh * 32 + segE + 8] = *(const uint4*)(g + 8);
#endif
  };
  auto stageA = [&](int buf, int kb) {
    if constexpr (sizeof(AT) == 2) {           // bf16 activations: pure copy
      const AT* g = A + (size_t)(m0 + rowh) * 1024 + kb * 32 + segE;
#if USE_ASYNC_LDS
      const unsigned l = lds_addr32(&As[buf][rowh * 32 + segE]);
      async_b128(l,      g);
      async_b128(l + 16, g + 8);
#else
      *(uint4*)&As[buf][rowh * 32 + segE]     = *(const uint4*)g;
      *(uint4*)&As[buf][rowh * 32 + segE + 8] = *(const uint4*)(g + 8);
#endif
    } else {                                    // fp32 activations: convert
      #pragma unroll
      for (int i = 0; i < 4; ++i) {
        const int row = rg + i * 32;
        const float4 v = *(const float4*)(A + (size_t)(m0 + row) * 1024 + kb * 32 + kq * 4);
        uint2 packed;
        packed.x = (unsigned)f32_bf16(v.x) | ((unsigned)f32_bf16(v.y) << 16);
        packed.y = (unsigned)f32_bf16(v.z) | ((unsigned)f32_bf16(v.w) << 16);
        *(uint2*)&As[buf][row * 32 + kq * 4] = packed;
      }
    }
  };

  stageB(0, 0);
  stageA(0, 0);
  wait_async0();
  __syncthreads();

  for (int kb = 0; kb < 1024 / 32; ++kb) {
    const int cur = kb & 1, nxt = cur ^ 1;
    if (kb + 1 < 32) {                          // fill next buffer while computing
      stageB(nxt, kb + 1);
      stageA(nxt, kb + 1);
    }

    Frag af[4], bfr[2];
    #pragma unroll
    for (int mi = 0; mi < 4; ++mi) {
      const int r = wm * 64 + mi * 16 + l16;
      af[mi].q[0] = *(const uint4*)&As[cur][r * 32 + hi * 8];
      af[mi].q[1] = *(const uint4*)&As[cur][r * 32 + hi * 8 + 16];
    }
    #pragma unroll
    for (int ni = 0; ni < 2; ++ni) {
      const int n = wn * 32 + ni * 16 + l16;
      bfr[ni].q[0] = *(const uint4*)&Bs[cur][n * 32 + hi * 8];
      bfr[ni].q[1] = *(const uint4*)&Bs[cur][n * 32 + hi * 8 + 16];
    }
    #pragma unroll
    for (int mi = 0; mi < 4; ++mi)
      #pragma unroll
      for (int ni = 0; ni < 2; ++ni)
        acc[mi][ni] = wmma_bf16(af[mi], bfr[ni], acc[mi][ni]);

    if (kb + 1 < 32) wait_async0();
    __syncthreads();
  }

  // epilogue: C/D layout lane=N, VGPR r -> M = r + 8*hi
  #pragma unroll
  for (int ni = 0; ni < 2; ++ni) {
    const int col = n0 + wn * 32 + ni * 16 + l16;
    const float bv = bias[col];
    #pragma unroll
    for (int mi = 0; mi < 4; ++mi) {
      #pragma unroll
      for (int r = 0; r < 8; ++r) {
        const int row = m0 + wm * 64 + mi * 16 + r + hi * 8;
        const float v = (acc[mi][ni][r] + bv) * outScale;
        if constexpr (MODE == 0) {
          ((float*)OutP)[(size_t)row * 1024 + col] = v;
        } else {
          const int bb = row >> 11, s = row & 2047;    // S = 2048
          const int h  = col >> 6,  d = col & 63;      // hd = 64
          const unsigned short bits = f32_bf16(v);
          if constexpr (MODE == 1)
            ((unsigned short*)OutP)[(((size_t)bb * 16 + h) * 2048 + s) * 64 + d] = bits;
          else
            ((unsigned short*)OutP)[(((size_t)bb * 16 + h) * 64 + d) * 2048 + s] = bits;
        }
      }
    }
  }
}

// ---------------------------------------------------------------------------
// Flash attention: grid (S/128, B*H), 8 waves/block, 16 query rows per wave.
// Q already carries the 1/sqrt(hd) scale; row max via v_permlane16; row sums
// via P x ones WMMA.
// ---------------------------------------------------------------------------
__global__ __launch_bounds__(256) void attn_k(const unsigned short* __restrict__ Q,
                                              const unsigned short* __restrict__ K,
                                              const unsigned short* __restrict__ Vt,
                                              unsigned short* __restrict__ Ctx) {
  __shared__ __align__(16) unsigned short Pl[8][16 * 32];  // per-wave P transpose tile

  const int tid  = threadIdx.x;
  const int lane = tid & 31, wv = tid >> 5;
  const int hi   = lane >> 4, l16 = lane & 15;
  const int bh   = blockIdx.y;
  const int b    = bh >> 4, h = bh & 15;
  const int q0   = blockIdx.x * 128 + wv * 16;

  // Q fragments held in registers for the whole key loop (A layout)
  Frag qf[2];
  const unsigned short* Qp = Q + ((size_t)bh * 2048 + q0 + l16) * 64;
  #pragma unroll
  for (int dt = 0; dt < 2; ++dt) {
    const int off = dt * 32 + hi * 16;
    qf[dt].q[0] = *(const uint4*)(Qp + off);
    qf[dt].q[1] = *(const uint4*)(Qp + off + 8);
  }

  // constant all-ones B fragment (bf16 1.0 = 0x3F80)
  Frag ones;
  { uint4 ov; ov.x = ov.y = ov.z = ov.w = 0x3F803F80u; ones.q[0] = ov; ones.q[1] = ov; }

  float rmax[8];
  v8f accO[4], accSum;
  #pragma unroll
  for (int r = 0; r < 8; ++r) rmax[r] = -3.0e38f;
  #pragma unroll
  for (int d = 0; d < 4; ++d) accO[d] = vzero();
  accSum = vzero();

  for (int kc = 0; kc < 2048 / 32; ++kc) {
    const int kBase = kc * 32;
    if (kc + 1 < 64) {          // warm caches for next chunk (global_prefetch_b8)
      __builtin_prefetch(K  + ((size_t)bh * 2048 + kBase + 32 + l16) * 64, 0, 1);
      __builtin_prefetch(Vt + ((size_t)bh * 64 + l16) * 2048 + kBase + 32, 0, 1);
    }

    // ---- scores: S = (Q*scale) * K^T ----
    v8f sAcc[2] = {vzero(), vzero()};
    #pragma unroll
    for (int nt = 0; nt < 2; ++nt) {
      #pragma unroll
      for (int dt = 0; dt < 2; ++dt) {
        Frag kf;
        const unsigned short* Kp =
            K + ((size_t)bh * 2048 + kBase + nt * 16 + l16) * 64 + dt * 32 + hi * 16;
        kf.q[0] = ((const uint4*)Kp)[0];
        kf.q[1] = ((const uint4*)Kp)[1];
        sAcc[nt] = wmma_bf16(qf[dt], kf, sAcc[nt]);
      }
    }

    // ---- online softmax: max via VALU permlane reduce; sums via WMMA below ----
    #pragma unroll
    for (int r = 0; r < 8; ++r) {
      const float a0 = sAcc[0][r], a1 = sAcc[1][r];
      const float m = max16(fmaxf(a0, a1));
      const float nm    = fmaxf(rmax[r], m);
      const float alpha = __expf(rmax[r] - nm);
      rmax[r] = nm;
      const float e0 = __expf(a0 - nm), e1 = __expf(a1 - nm);
      accSum[r] *= alpha;
      #pragma unroll
      for (int d = 0; d < 4; ++d) accO[d][r] *= alpha;
      const int pr = r + hi * 8;                 // C layout -> LDS row-major P
      Pl[wv][pr * 32 + l16]      = f32_bf16(e0);
      Pl[wv][pr * 32 + 16 + l16] = f32_bf16(e1);
    }
    asm volatile("s_wait_dscnt 0x0" ::: "memory"); // wave-local DS RAW ordering

    // ---- P back as A-fragment (transpose via LDS) ----
    Frag pf;
    pf.q[0] = *(const uint4*)&Pl[wv][l16 * 32 + hi * 16];
    pf.q[1] = *(const uint4*)&Pl[wv][l16 * 32 + hi * 16 + 8];

    // row sums on the matrix unit: accSum += P * 1
    accSum = wmma_bf16(pf, ones, accSum);

    // ---- O += P * V ----
    #pragma unroll
    for (int dt = 0; dt < 4; ++dt) {
      Frag vf;
      const unsigned short* Vp =
          Vt + ((size_t)bh * 64 + dt * 16 + l16) * 2048 + kBase + hi * 16;
      vf.q[0] = ((const uint4*)Vp)[0];
      vf.q[1] = ((const uint4*)Vp)[1];
      accO[dt] = wmma_bf16(pf, vf, accO[dt]);
    }
  }

  // ---- normalize and emit ctx bf16 [B,S,H,hd] ----
  #pragma unroll
  for (int r = 0; r < 8; ++r) {
    const float inv = 1.0f / accSum[r];
    const int row = q0 + r + hi * 8;
    #pragma unroll
    for (int dt = 0; dt < 4; ++dt) {
      const float v = accO[dt][r] * inv;
      Ctx[((size_t)b * 2048 + row) * 1024 + h * 64 + dt * 16 + l16] = f32_bf16(v);
    }
  }
}

// ---------------------------------------------------------------------------
extern "C" void kernel_launch(void* const* d_in, const int* in_sizes, int n_in,
                              void* d_out, int out_size, void* d_ws, size_t ws_size,
                              hipStream_t stream) {
  (void)in_sizes; (void)n_in; (void)out_size; (void)ws_size;
  const float* q  = (const float*)d_in[0];
  const float* k  = (const float*)d_in[1];
  const float* v  = (const float*)d_in[2];
  const float* Wq = (const float*)d_in[3];
  const float* bq = (const float*)d_in[4];
  const float* Wk = (const float*)d_in[5];
  const float* bk = (const float*)d_in[6];
  const float* Wv = (const float*)d_in[7];
  const float* bv = (const float*)d_in[8];
  const float* Wo = (const float*)d_in[9];
  const float* bo = (const float*)d_in[10];

  char* ws = (char*)d_ws;
  const size_t SEG = (size_t)2 * 16 * 2048 * 64 * sizeof(unsigned short);  // 8 MiB
  const size_t WSEG = (size_t)1024 * 1024 * sizeof(unsigned short);        // 2 MiB
  unsigned short* Qb  = (unsigned short*)(ws);
  unsigned short* Kb  = (unsigned short*)(ws + SEG);
  unsigned short* Vtb = (unsigned short*)(ws + 2 * SEG);
  unsigned short* Cb  = (unsigned short*)(ws + 3 * SEG);
  unsigned short* Wtq = (unsigned short*)(ws + 4 * SEG);
  unsigned short* Wtk = (unsigned short*)(ws + 4 * SEG + WSEG);
  unsigned short* Wtv = (unsigned short*)(ws + 4 * SEG + 2 * WSEG);
  unsigned short* Wto = (unsigned short*)(ws + 4 * SEG + 3 * WSEG);

  const dim3 blk(256), gt(16, 16), gg(8, 32);
  transp_k<<<gt, blk, 0, stream>>>(Wq, Wtq);
  transp_k<<<gt, blk, 0, stream>>>(Wk, Wtk);
  transp_k<<<gt, blk, 0, stream>>>(Wv, Wtv);
  transp_k<<<gt, blk, 0, stream>>>(Wo, Wto);

  gemm_k<1, float><<<gg, blk, 0, stream>>>(q, Wtq, bq, Qb, 0.125f);  // fold 1/sqrt(hd)
  gemm_k<1, float><<<gg, blk, 0, stream>>>(k, Wtk, bk, Kb, 1.0f);
  gemm_k<2, float><<<gg, blk, 0, stream>>>(v, Wtv, bv, Vtb, 1.0f);
  attn_k<<<dim3(16, 32), blk, 0, stream>>>(Qb, Kb, Vtb, Cb);
  gemm_k<0, unsigned short><<<gg, blk, 0, stream>>>(Cb, Wto, bo, d_out, 1.0f);
}
